// NeuralAttention_45329084842553
// MI455X (gfx1250) — matrix operand order repaired and checked by reading, not compiled, not run on hardware
//
#include <hip/hip_runtime.h>

// Problem constants (match reference)
#define BB   8
#define TT   1024
#define HIDN 1024
#define NH   16
#define HD   64

typedef __attribute__((ext_vector_type(16))) __bf16 v16bf;
typedef __attribute__((ext_vector_type(8)))  __bf16 v8bf;
typedef __attribute__((ext_vector_type(8)))  float  v8f;
typedef __attribute__((ext_vector_type(4)))  unsigned int v4u;
typedef __attribute__((ext_vector_type(8)))  unsigned int v8u;

static __device__ __forceinline__ v16bf cat8(v8bf lo, v8bf hi) {
  return __builtin_shufflevector(lo, hi, 0,1,2,3,4,5,6,7,8,9,10,11,12,13,14,15);
}

// ---------------------------------------------------------------------------
// TDM: 2-D tensor_load_to_lds of a 32-row x 64-col bf16 tile whose rows are
// strided by HIDN elements in global memory, packed contiguously into LDS.
// Descriptor per CDNA5 ISA (group0) / §8.4 (group1).
// 2-D tensor -> only descriptor groups 0 and 1 (VADDR2/3 omitted).
// Tracked by TENSORcnt; EXEC is ignored (one DMA per wave).
// ---------------------------------------------------------------------------
static __device__ __forceinline__ void tdm_load_v_tile(unsigned lds_off,
                                                       const __bf16* gptr) {
  const unsigned long long ga = (unsigned long long)gptr;
  v4u g0;
  g0[0] = 1u;                                   // count=1, user-mode, no gather
  g0[1] = lds_off;                              // lds_addr (bytes)
  g0[2] = (unsigned)ga;                         // global_addr[31:0]
  g0[3] = (unsigned)((ga >> 32) & 0x01ffffffu)  // global_addr[56:32]
          | (2u << 30);                         // type=2 ("image")
  v8u g1;
  g1[0] = 1u << 16;          // wg_mask=0 | data_size=1 (2 bytes/elem)
  g1[1] = 64u << 16;         // atomic_barrier_addr=0 | tensor_dim0[15:0]=64
  g1[2] = (unsigned)TT << 16;// tensor_dim0[31:16]=0 | tensor_dim1[15:0]=1024
  g1[3] = 64u << 16;         // tensor_dim1[31:16]=0 | tile_dim0=64
  g1[4] = 32u;               // tile_dim1=32 rows | tile_dim2=0
  g1[5] = (unsigned)HIDN;    // tensor_dim0_stride[31:0]=1024 elems
  g1[6] = 0u;                // stride[47:32]=0 | tensor_dim1_stride lo=0
  g1[7] = 0u;
  asm volatile("tensor_load_to_lds %0, %1" :: "s"(g0), "s"(g1) : "memory");
}

// ---------------------------------------------------------------------------
// Kernel 1: f32 -> bf16 conversion (vectorized x4)
// ---------------------------------------------------------------------------
__global__ void cvt_f32_to_bf16(const float* __restrict__ src,
                                __bf16* __restrict__ dst, int n) {
  int i4 = (blockIdx.x * blockDim.x + threadIdx.x) * 4;
  if (i4 + 3 < n) {
    float4 v = *(const float4*)(src + i4);
    dst[i4 + 0] = (__bf16)v.x;
    dst[i4 + 1] = (__bf16)v.y;
    dst[i4 + 2] = (__bf16)v.z;
    dst[i4 + 3] = (__bf16)v.w;
  }
}

// ---------------------------------------------------------------------------
// Kernel 2: Y[M,N] = A[M,K] @ W[N,K]^T  (bf16 in, f32 accumulate via WMMA)
// 256 threads = 8 waves; wave tile 32x32 (2x2 WMMA), block tile 64x128.
// ---------------------------------------------------------------------------
template <bool OUT_F32>
__global__ __launch_bounds__(256)
void gemm_xwt(const __bf16* __restrict__ A, const __bf16* __restrict__ W,
              float* __restrict__ outF, __bf16* __restrict__ outB,
              int M, int N, int K) {
  const int lane  = threadIdx.x & 31;
  const int wave  = threadIdx.x >> 5;
  const int waveM = wave >> 2;
  const int waveN = wave & 3;
  const int m0    = blockIdx.x * 64  + waveM * 32;
  const int n0    = blockIdx.y * 128 + waveN * 32;
  const int lrow  = lane & 15;
  const int lhalf = lane >> 4;

  v8f acc[2][2] = {};

  for (int k0 = 0; k0 < K; k0 += 32) {
    v16bf a[2], b[2];
#pragma unroll
    for (int mi = 0; mi < 2; ++mi) {
      const __bf16* pa = A + (size_t)(m0 + mi * 16 + lrow) * K + k0 + lhalf * 16;
      a[mi] = cat8(*(const v8bf*)pa, *(const v8bf*)(pa + 8));
      if (mi == 0 && k0 + 32 < K) __builtin_prefetch(pa + 32, 0, 0);
    }
#pragma unroll
    for (int ni = 0; ni < 2; ++ni) {
      const __bf16* pb = W + (size_t)(n0 + ni * 16 + lrow) * K + k0 + lhalf * 16;
      b[ni] = *(const v16bf*)pb;
      if (ni == 0 && k0 + 32 < K) __builtin_prefetch(pb + 32, 0, 0);
    }
#pragma unroll
    for (int mi = 0; mi < 2; ++mi)
#pragma unroll
      for (int ni = 0; ni < 2; ++ni)
        acc[mi][ni] = __builtin_amdgcn_wmma_f32_16x16x32_bf16(
            false, a[mi], false, b[ni], (short)0, acc[mi][ni], false, false);
  }

#pragma unroll
  for (int mi = 0; mi < 2; ++mi) {
    const int rbase = m0 + mi * 16 + lhalf * 8;
#pragma unroll
    for (int ni = 0; ni < 2; ++ni) {
      const int col = n0 + ni * 16 + lrow;
#pragma unroll
      for (int r = 0; r < 8; ++r) {
        const size_t idx = (size_t)(rbase + r) * N + col;
        if constexpr (OUT_F32) outF[idx] = acc[mi][ni][r];
        else                   outB[idx] = (__bf16)acc[mi][ni][r];
      }
    }
  }
}

// ---------------------------------------------------------------------------
// Kernel 3: RoPE on q and k (bf16 in-place), timestamp gather.
// ---------------------------------------------------------------------------
__global__ void rope_qk(__bf16* __restrict__ q, __bf16* __restrict__ k,
                        const float* __restrict__ cosT,
                        const float* __restrict__ sinT,
                        const int* __restrict__ ts) {
  const int i = blockIdx.x * blockDim.x + threadIdx.x;  // 2^22 threads
  const int d = i & 31;
  const int h = (i >> 5) & (NH - 1);
  const int t = (i >> 9) & (TT - 1);
  const int b = i >> 19;

  const int tp = ts[b * TT + t];
  const float c1 = cosT[tp * HD + d],      s1 = sinT[tp * HD + d];
  const float c2 = cosT[tp * HD + d + 32], s2 = sinT[tp * HD + d + 32];

  const size_t base = (size_t)(b * TT + t) * HIDN + h * HD;
  const float q1 = (float)q[base + d], q2 = (float)q[base + d + 32];
  q[base + d]      = (__bf16)(q1 * c1 - q2 * s1);
  q[base + d + 32] = (__bf16)(q2 * c2 + q1 * s2);
  const float k1 = (float)k[base + d], k2 = (float)k[base + d + 32];
  k[base + d]      = (__bf16)(k1 * c1 - k2 * s1);
  k[base + d + 32] = (__bf16)(k2 * c2 + k1 * s2);
}

// ---------------------------------------------------------------------------
// Kernel 4: attention for one (b, h, 64-query tile). 4 waves x 16 q-rows.
// LDS: 4 x 64KB f32 score strips + row sums + 4 x (2 x 4KB) TDM-staged V
// double buffers = ~288KB of the 320KB WGP LDS.
// V tiles arrive via tensor_load_to_lds (TENSORcnt double-buffered), so the
// P*V B-fragments are built from LDS instead of strided HBM gathers, and
// DMA of tile i+1 overlaps WMMA on tile i.
// ---------------------------------------------------------------------------
__global__ __launch_bounds__(128)
void attn_fwd(const __bf16* __restrict__ q, const __bf16* __restrict__ k,
              const __bf16* __restrict__ v, const int* __restrict__ mask,
              __bf16* __restrict__ ctx) {
  extern __shared__ char smem[];
  const int lane  = threadIdx.x & 31;
  const int wave  = threadIdx.x >> 5;
  const int h     = blockIdx.y;
  const int b     = blockIdx.z;
  const int q0    = blockIdx.x * 64 + wave * 16;
  const int lrow  = lane & 15;
  const int lhalf = lane >> 4;

  float* S    = (float*)smem + (size_t)wave * 16 * TT;   // wave-private 16x1024
  float* rinv = (float*)smem + (size_t)4 * 16 * TT + wave * 16;
  const unsigned VOFF = 4u * 16u * TT * 4u + 256u;       // after S + rinv
  const unsigned myV  = VOFF + (unsigned)wave * 2u * 4096u;

  // Q fragments: 16 rows x 64 head-dims = 2 WMMA k-chunks, kept in VGPRs
  v16bf aq[2];
#pragma unroll
  for (int c = 0; c < 2; ++c) {
    const __bf16* pq =
        q + (size_t)(b * TT + q0 + lrow) * HIDN + h * HD + c * 32 + lhalf * 16;
    aq[c] = cat8(*(const v8bf*)pq, *(const v8bf*)(pq + 8));
  }

  // ---- S = scale * Q K^T, masked, into LDS ----
  for (int kt = 0; kt < TT; kt += 16) {
    v8f s = {};
#pragma unroll
    for (int c = 0; c < 2; ++c) {
      const __bf16* pk =
          k + (size_t)(b * TT + kt + lrow) * HIDN + h * HD + c * 32 + lhalf * 16;
      v16bf bk = *(const v16bf*)pk;
      s = __builtin_amdgcn_wmma_f32_16x16x32_bf16(false, aq[c], false, bk,
                                                  (short)0, s, false, false);
    }
    const int key = kt + lrow;
    const int rb  = lhalf * 8;
#pragma unroll
    for (int r = 0; r < 8; ++r) {
      const int qrow = q0 + rb + r;
      float sc = s[r] * 0.125f;  // 1/sqrt(64)
      const int mv = mask[((size_t)b * TT + qrow) * TT + key];
      S[(rb + r) * TT + key] = mv ? sc : -3.0e38f;
    }
  }

  // ---- softmax over each of this wave's 16 rows (32 lanes cooperate) ----
  for (int r = 0; r < 16; ++r) {
    float* row = S + r * TT;
    float mx = -3.4e38f;
    for (int i = lane; i < TT; i += 32) mx = fmaxf(mx, row[i]);
#pragma unroll
    for (int off = 16; off > 0; off >>= 1) mx = fmaxf(mx, __shfl_xor(mx, off, 32));
    float sum = 0.f;
    for (int i = lane; i < TT; i += 32) {
      float e = __expf(row[i] - mx);
      row[i] = e;
      sum += e;
    }
#pragma unroll
    for (int off = 16; off > 0; off >>= 1) sum += __shfl_xor(sum, off, 32);
    if (lane == 0) rinv[r] = 1.0f / sum;
  }

  // ---- O = (P / rowsum) @ V, V tiles staged by TDM double-buffer ----
  const float rs = rinv[lrow];
  const __bf16* vbase = v + (size_t)b * TT * HIDN + h * HD;  // rows stride HIDN
  constexpr int NIT = TT / 32;

  tdm_load_v_tile(myV, vbase);  // tile 0 -> buffer 0

  v8f o[4] = {};
  for (int it = 0; it < NIT; ++it) {
    const int kc = it * 32;
    if (it + 1 < NIT) {
      tdm_load_v_tile(myV + (unsigned)((it + 1) & 1) * 4096u,
                      vbase + (size_t)(kc + 32) * HIDN);
      __builtin_amdgcn_s_wait_tensorcnt(1);  // tile `it` has landed
    } else {
      __builtin_amdgcn_s_wait_tensorcnt(0);
    }
    const __bf16* vt = (const __bf16*)(smem + myV + (unsigned)(it & 1) * 4096u);

    const float* prow = S + lrow * TT + kc + lhalf * 16;
    v16bf ap;
#pragma unroll
    for (int j = 0; j < 8; ++j) ap[j] = (__bf16)(prow[j] * rs);
#pragma unroll
    for (int j = 0; j < 8; ++j) ap[8 + j] = (__bf16)(prow[8 + j] * rs);

#pragma unroll
    for (int ni = 0; ni < 4; ++ni) {
      const int d = ni * 16 + lrow;
      v16bf bv;  // B-fragment: lane = column d, 16 k-rows from staged tile
#pragma unroll
      for (int j = 0; j < 16; ++j) bv[j] = vt[(lhalf * 16 + j) * HD + d];
      o[ni] = __builtin_amdgcn_wmma_f32_16x16x32_bf16(false, ap, false, bv,
                                                      (short)0, o[ni], false, false);
    }
  }

  const int rb = lhalf * 8;
#pragma unroll
  for (int ni = 0; ni < 4; ++ni)
#pragma unroll
    for (int r = 0; r < 8; ++r)
      ctx[(size_t)(b * TT + q0 + rb + r) * HIDN + h * HD + ni * 16 + lrow] =
          (__bf16)o[ni][r];
}

// ---------------------------------------------------------------------------
// Launch: convert -> QKV GEMMs -> RoPE -> attention -> output GEMM
// ---------------------------------------------------------------------------
extern "C" void kernel_launch(void* const* d_in, const int* in_sizes, int n_in,
                              void* d_out, int out_size, void* d_ws, size_t ws_size,
                              hipStream_t stream) {
  const float* x    = (const float*)d_in[0];
  const float* Wq   = (const float*)d_in[1];
  const float* Wk   = (const float*)d_in[2];
  const float* Wv   = (const float*)d_in[3];
  const float* Wo   = (const float*)d_in[4];
  const float* cosT = (const float*)d_in[5];
  const float* sinT = (const float*)d_in[6];
  const int*   mask = (const int*)d_in[7];
  const int*   ts   = (const int*)d_in[8];
  float* out = (float*)d_out;

  const size_t MT = (size_t)BB * TT;  // 8192 rows
  char* ws = (char*)d_ws;
  size_t off = 0;
  auto alloc = [&](size_t bytes) {
    char* p = ws + off;
    off += (bytes + 255) & ~(size_t)255;
    return p;
  };
  __bf16* xb  = (__bf16*)alloc(MT * HIDN * sizeof(__bf16));
  __bf16* wqb = (__bf16*)alloc((size_t)HIDN * HIDN * sizeof(__bf16));
  __bf16* wkb = (__bf16*)alloc((size_t)HIDN * HIDN * sizeof(__bf16));
  __bf16* wvb = (__bf16*)alloc((size_t)HIDN * HIDN * sizeof(__bf16));
  __bf16* wob = (__bf16*)alloc((size_t)HIDN * HIDN * sizeof(__bf16));
  __bf16* qb  = (__bf16*)alloc(MT * HIDN * sizeof(__bf16));
  __bf16* kb  = (__bf16*)alloc(MT * HIDN * sizeof(__bf16));
  __bf16* vb  = (__bf16*)alloc(MT * HIDN * sizeof(__bf16));
  __bf16* cb  = (__bf16*)alloc(MT * HIDN * sizeof(__bf16));
  (void)ws_size; (void)in_sizes; (void)n_in; (void)out_size;

  const int thr = 256;
  cvt_f32_to_bf16<<<(int)(MT * HIDN / 4 / thr), thr, 0, stream>>>(x, xb, (int)(MT * HIDN));
  cvt_f32_to_bf16<<<HIDN * HIDN / 4 / thr, thr, 0, stream>>>(Wq, wqb, HIDN * HIDN);
  cvt_f32_to_bf16<<<HIDN * HIDN / 4 / thr, thr, 0, stream>>>(Wk, wkb, HIDN * HIDN);
  cvt_f32_to_bf16<<<HIDN * HIDN / 4 / thr, thr, 0, stream>>>(Wv, wvb, HIDN * HIDN);
  cvt_f32_to_bf16<<<HIDN * HIDN / 4 / thr, thr, 0, stream>>>(Wo, wob, HIDN * HIDN);

  dim3 gg((unsigned)(MT / 64), HIDN / 128);
  gemm_xwt<false><<<gg, 256, 0, stream>>>(xb, wqb, nullptr, qb, (int)MT, HIDN, HIDN);
  gemm_xwt<false><<<gg, 256, 0, stream>>>(xb, wkb, nullptr, kb, (int)MT, HIDN, HIDN);
  gemm_xwt<false><<<gg, 256, 0, stream>>>(xb, wvb, nullptr, vb, (int)MT, HIDN, HIDN);

  rope_qk<<<(int)(MT * NH * 32 / 256), 256, 0, stream>>>(qb, kb, cosT, sinT, ts);

  const size_t smem = (size_t)4 * 16 * TT * sizeof(float) + 256 +
                      (size_t)4 * 2 * 4096;  // scores + rinv + V buffers
  attn_fwd<<<dim3(TT / 64, NH, BB), 128, smem, stream>>>(qb, kb, vb, mask, cb);

  gemm_xwt<true><<<gg, 256, 0, stream>>>(cb, wob, out, nullptr, (int)MT, HIDN, HIDN);
}
